// MultiHeadSelfAttention_29016799052072
// MI455X (gfx1250) — compile-verified, matrix operand order checked by reading, not run on hardware
//
#include <hip/hip_runtime.h>

// ---------------- problem constants ----------------
constexpr int kB  = 4;
constexpr int kT  = 2048;
constexpr int kD  = 1024;
constexpr int kH  = 16;
constexpr int kHD = 64;
constexpr int kBH = kB * kH;   // 64
constexpr int kM  = kB * kT;   // 8192 rows of x / y

typedef unsigned short u16;
typedef unsigned int   u32;
typedef __attribute__((ext_vector_type(16))) __bf16 v16bf;
typedef __attribute__((ext_vector_type(8)))  float  v8f;
typedef __attribute__((ext_vector_type(4)))  u32    u32x4;

union FragBF { v16bf v; u32x4 q[2]; };   // 32 bytes: one 16x32 bf16 A/B operand
union AccF   { v8f   v; float f[8];  };  // 32 bytes: one 16x16 f32 C/D tile

__device__ __forceinline__ u16 f2bf(float f) {  // RNE float->bf16
  u32 u = __float_as_uint(f);
  u32 r = u + 0x7FFFu + ((u >> 16) & 1u);
  return (u16)(r >> 16);
}

__device__ __forceinline__ v8f wmma_bf16(const v16bf& a, const v16bf& b, const v8f& c) {
  return __builtin_amdgcn_wmma_f32_16x16x32_bf16(false, a, false, b, (short)0, c, false, false);
}

// Async DMA: 16 bytes global -> LDS, tracked by ASYNCcnt.
// LDS aperture: generic addr low 32 bits == LDS byte offset.
__device__ __forceinline__ void async_cp16(u16* ldsDst, const u16* gSrc) {
  u32 ldsOff = (u32)(size_t)ldsDst;
  asm volatile("global_load_async_to_lds_b128 %0, %1, off"
               :: "v"(ldsOff), "v"(gSrc) : "memory");
}
#define WAIT_ASYNC0() asm volatile("s_wait_asynccnt 0x0" ::: "memory")
#define WAIT_DS0()    asm volatile("s_wait_dscnt 0x0"    ::: "memory")

// ---------------- packing kernels ----------------
__global__ void pack_bf16_kernel(const float* __restrict__ src,
                                 u16* __restrict__ dst, int n) {
  int i = blockIdx.x * blockDim.x + threadIdx.x;
  int stride = gridDim.x * blockDim.x;
  for (; i < n; i += stride) dst[i] = f2bf(src[i]);
}

// w [K,N] f32 row-major -> wT [N,K] bf16 row-major (so GEMM B-frag reads are contiguous in K)
__global__ void transpose_pack_kernel(const float* __restrict__ w,
                                      u16* __restrict__ wT, int K, int N) {
  int i = blockIdx.x * blockDim.x + threadIdx.x;
  int total = K * N;
  int stride = gridDim.x * blockDim.x;
  for (; i < total; i += stride) {
    int k = i / N, n = i % N;
    wT[(size_t)n * K + k] = f2bf(w[i]);
  }
}

// ---------------- bf16 GEMM: C[M,N] = A[M,K] * BT[N,K]^T + bias ----------------
// mode 0: N==3*kD, scatter bf16 to q(*1/8) [BH,T,HD], k [BH,T,HD], vT [BH,HD,T]
// mode 1: N==kD,  write fp32 to outf [M,N]
// Double-buffered LDS staging via GLOBAL_LOAD_ASYNC_TO_LDS_B128.
__global__ __launch_bounds__(256) void gemm_bf16_kernel(
    const u16* __restrict__ A, const u16* __restrict__ BT,
    const float* __restrict__ bias, int K, int N, int mode,
    u16* __restrict__ qOut, u16* __restrict__ kOut, u16* __restrict__ vTOut,
    float* __restrict__ outf)
{
  // 128x32 tiles, row stride 40 elems (80B) -> conflict-free b128 fragment reads
  __shared__ __align__(16) u16 lAbuf[2][128 * 40];
  __shared__ __align__(16) u16 lBbuf[2][128 * 40];

  const int tid  = threadIdx.x;
  const int wave = tid >> 5;
  const int lane = tid & 31;
  const int hi   = lane >> 4;   // selects K sub-chunk per ISA 16-bit operand layout
  const int ln   = lane & 15;
  const int wm   = wave & 3;    // 4 waves along M  -> 32 rows each
  const int wn   = wave >> 2;   // 2 waves along N  -> 64 cols each
  const int mBase = blockIdx.x * 128;
  const int nBase = blockIdx.y * 128;

  AccF acc[2][4];
#pragma unroll
  for (int mt = 0; mt < 2; ++mt)
#pragma unroll
    for (int nt = 0; nt < 4; ++nt)
#pragma unroll
      for (int r = 0; r < 8; ++r) acc[mt][nt].f[r] = 0.0f;

  const int srow = tid >> 2;        // 0..63
  const int sc8  = (tid & 3) * 8;   // element chunk within 32-wide K slab

  // issue async staging of one 128x32 A slab + 128x32 B slab into buffer `buf`
  auto stage_async = [&](int buf, int kb) {
#pragma unroll
    for (int r0 = 0; r0 < 128; r0 += 64) {
      int row = r0 + srow;
      async_cp16(&lAbuf[buf][row * 40 + sc8],
                 A  + (size_t)(mBase + row) * K + kb + sc8);
      async_cp16(&lBbuf[buf][row * 40 + sc8],
                 BT + (size_t)(nBase + row) * K + kb + sc8);
    }
  };

  stage_async(0, 0);
  int it = 0;
  for (int kb = 0; kb < K; kb += 32, ++it) {
    const int buf = it & 1;
    WAIT_ASYNC0();        // tile `it` landed in LDS (only outstanding async ops)
    __syncthreads();      // all waves done with buffer buf^1 from iteration it-1
    if (kb + 32 < K) stage_async(buf ^ 1, kb + 32);   // DMA next tile during compute
    if (kb + 64 < K) {    // pull the slab after that toward the WGP caches
      __builtin_prefetch(A  + (size_t)(mBase + srow) * K + kb + 64 + sc8, 0, 3);
      __builtin_prefetch(BT + (size_t)(nBase + srow) * K + kb + 64 + sc8, 0, 3);
    }

    const u16* la = lAbuf[buf];
    const u16* lb = lBbuf[buf];
    FragBF af[2], bfr[4];
#pragma unroll
    for (int mt = 0; mt < 2; ++mt) {
      int row = wm * 32 + mt * 16 + ln;
      af[mt].q[0] = *(const u32x4*)&la[row * 40 + hi * 8];
      af[mt].q[1] = *(const u32x4*)&la[row * 40 + 16 + hi * 8];
    }
#pragma unroll
    for (int nt = 0; nt < 4; ++nt) {
      int row = wn * 64 + nt * 16 + ln;
      bfr[nt].q[0] = *(const u32x4*)&lb[row * 40 + hi * 8];
      bfr[nt].q[1] = *(const u32x4*)&lb[row * 40 + 16 + hi * 8];
    }
#pragma unroll
    for (int mt = 0; mt < 2; ++mt)
#pragma unroll
      for (int nt = 0; nt < 4; ++nt)
        acc[mt][nt].v = wmma_bf16(af[mt].v, bfr[nt].v, acc[mt][nt].v);
  }

  // Epilogue. C/D element (m,n): lane = n%16 + 16*(m/8), vgpr = m%8.
#pragma unroll
  for (int mt = 0; mt < 2; ++mt) {
#pragma unroll
    for (int nt = 0; nt < 4; ++nt) {
      int n = nBase + wn * 64 + nt * 16 + ln;
      float bv = bias[n];
#pragma unroll
      for (int r = 0; r < 8; ++r) {
        int m = mBase + wm * 32 + mt * 16 + r + 8 * hi;
        float val = acc[mt][nt].f[r] + bv;
        if (mode == 0) {
          int g  = n >> 10;          // 0=q 1=k 2=v
          int h  = (n >> 6) & 15;
          int hd = n & 63;
          int b  = m >> 11;
          int t  = m & (kT - 1);
          int bh = b * kH + h;
          if (g == 0)
            qOut[((size_t)bh * kT + t) * kHD + hd] = f2bf(val * 0.125f); // fold 1/sqrt(64)
          else if (g == 1)
            kOut[((size_t)bh * kT + t) * kHD + hd] = f2bf(val);
          else
            vTOut[((size_t)bh * kHD + hd) * kT + t] = f2bf(val);        // V transposed
        } else {
          outf[(size_t)m * N + n] = val;
        }
      }
    }
  }
}

// ---------------- causal flash attention ----------------
// Q,K: [BH,T,HD] bf16 (Q pre-scaled by 1/sqrt(HD)); VT: [BH,HD,T] bf16
// Y: [B,T,H,HD] bf16. Block = 8 waves; each wave owns 16 q rows (128/block).
// K/VT tiles double-buffered in LDS via async DMA.
__global__ __launch_bounds__(256) void flash_attn_kernel(
    const u16* __restrict__ Q, const u16* __restrict__ Kb,
    const u16* __restrict__ VT, u16* __restrict__ Y)
{
  __shared__ __align__(16) u16 lKbuf[2][32 * 72];    // 32 keys x 64 hd, padded stride
  __shared__ __align__(16) u16 lVTbuf[2][64 * 40];   // 64 hd  x 32 keys, padded stride
  __shared__ __align__(16) u16 lP[8][16 * 40];       // per-wave P scratch 16x32

  const int tid  = threadIdx.x;
  const int wave = tid >> 5;
  const int lane = tid & 31;
  const int hi   = lane >> 4;
  const int ln   = lane & 15;
  const int bh     = blockIdx.y;
  const int qBlock = blockIdx.x * 128;
  const int qRow   = qBlock + wave * 16;
  const size_t qkBase = (size_t)bh * kT * kHD;
  const size_t vtBase = (size_t)bh * kHD * kT;

  // Q A-fragments (hd 0..31 and 32..63), resident in VGPRs for whole kernel
  FragBF qf[2];
  {
    const u16* qp = Q + qkBase + (size_t)(qRow + ln) * kHD;
    qf[0].q[0] = *(const u32x4*)(qp + hi * 8);
    qf[0].q[1] = *(const u32x4*)(qp + 16 + hi * 8);
    qf[1].q[0] = *(const u32x4*)(qp + 32 + hi * 8);
    qf[1].q[1] = *(const u32x4*)(qp + 48 + hi * 8);
  }

  AccF o[4];
#pragma unroll
  for (int ht = 0; ht < 4; ++ht)
#pragma unroll
    for (int r = 0; r < 8; ++r) o[ht].f[r] = 0.0f;
  float mrow[8], lrow[8];
#pragma unroll
  for (int r = 0; r < 8; ++r) { mrow[r] = -1.0e30f; lrow[r] = 0.0f; }

  const int kEnd     = qBlock + 128;   // block-uniform causal bound
  const int qMaxWave = qRow + 15;
  const float L2E = 1.44269504f;

  const int krow = tid >> 3;          // K stage: 0..31
  const int kc8  = (tid & 7) * 8;
  const int vrow = tid >> 2;          // VT stage: 0..63
  const int vc8  = (tid & 3) * 8;

  auto stage_async = [&](int buf, int kb) {
    async_cp16(&lKbuf[buf][krow * 72 + kc8],
               Kb + qkBase + (size_t)(kb + krow) * kHD + kc8);
    async_cp16(&lVTbuf[buf][vrow * 40 + vc8],
               VT + vtBase + (size_t)vrow * kT + kb + vc8);
  };

  stage_async(0, 0);
  int it = 0;
  for (int kb = 0; kb < kEnd; kb += 32, ++it) {
    const int buf = it & 1;
    WAIT_ASYNC0();        // current tile landed
    __syncthreads();      // everyone finished reading buffer buf^1
    if (kb + 32 < kEnd) stage_async(buf ^ 1, kb + 32);

    if (kb <= qMaxWave) {   // wave-uniform: skip fully-masked key blocks
      const u16* lk  = lKbuf[buf];
      const u16* lvt = lVTbuf[buf];
      // S = Q * K^T for 32 keys (2 key tiles x 2 hd halves)
      AccF s[2];
#pragma unroll
      for (int kt = 0; kt < 2; ++kt) {
#pragma unroll
        for (int r = 0; r < 8; ++r) s[kt].f[r] = 0.0f;
        int row = kt * 16 + ln;
        FragBF kf;
        kf.q[0] = *(const u32x4*)&lk[row * 72 + hi * 8];
        kf.q[1] = *(const u32x4*)&lk[row * 72 + 16 + hi * 8];
        s[kt].v = wmma_bf16(qf[0].v, kf.v, s[kt].v);
        kf.q[0] = *(const u32x4*)&lk[row * 72 + 32 + hi * 8];
        kf.q[1] = *(const u32x4*)&lk[row * 72 + 48 + hi * 8];
        s[kt].v = wmma_bf16(qf[1].v, kf.v, s[kt].v);
      }

      // causal mask + online softmax (rows r+8*hi, key cols kb+ln / kb+16+ln)
#pragma unroll
      for (int r = 0; r < 8; ++r) {
        int q = qRow + r + 8 * hi;
        if (kb + ln > q)       s[0].f[r] = -3.0e38f;
        if (kb + 16 + ln > q)  s[1].f[r] = -3.0e38f;
        float mx = fmaxf(s[0].f[r], s[1].f[r]);
#pragma unroll
        for (int d = 1; d < 16; d <<= 1) mx = fmaxf(mx, __shfl_xor(mx, d, 32));
        float mnew = fmaxf(mrow[r], mx);
        float corr = exp2f((mrow[r] - mnew) * L2E);
        float p0 = exp2f((s[0].f[r] - mnew) * L2E);
        float p1 = exp2f((s[1].f[r] - mnew) * L2E);
        lP[wave][(r + 8 * hi) * 40 + ln]      = f2bf(p0);
        lP[wave][(r + 8 * hi) * 40 + 16 + ln] = f2bf(p1);
#pragma unroll
        for (int ht = 0; ht < 4; ++ht) o[ht].f[r] *= corr;
        float ps = p0 + p1;
#pragma unroll
        for (int d = 1; d < 16; d <<= 1) ps += __shfl_xor(ps, d, 32);
        lrow[r] = lrow[r] * corr + ps;
        mrow[r] = mnew;
      }

      // lP writes are cross-lane within this wave; LDS is in-order per wave.
      WAIT_DS0();

      FragBF pf;
      pf.q[0] = *(const u32x4*)&lP[wave][ln * 40 + hi * 8];
      pf.q[1] = *(const u32x4*)&lP[wave][ln * 40 + 16 + hi * 8];
#pragma unroll
      for (int ht = 0; ht < 4; ++ht) {
        int row = ht * 16 + ln;
        FragBF vf;
        vf.q[0] = *(const u32x4*)&lvt[row * 40 + hi * 8];
        vf.q[1] = *(const u32x4*)&lvt[row * 40 + 16 + hi * 8];
        o[ht].v = wmma_bf16(pf.v, vf.v, o[ht].v);
      }
    }
  }

  // normalize and store Y[b,t,h,hd] as bf16 (row layout matches final GEMM A)
  const int b = bh >> 4, h = bh & 15;
#pragma unroll
  for (int r = 0; r < 8; ++r) {
    float inv = 1.0f / lrow[r];
    int t = qRow + r + 8 * hi;
    u16* yp = Y + ((size_t)(b * kT + t) * kH + h) * kHD;
#pragma unroll
    for (int ht = 0; ht < 4; ++ht)
      yp[ht * 16 + ln] = f2bf(o[ht].f[r] * inv);
  }
}

// ---------------- host launcher ----------------
extern "C" void kernel_launch(void* const* d_in, const int* in_sizes, int n_in,
                              void* d_out, int out_size, void* d_ws, size_t ws_size,
                              hipStream_t stream) {
  (void)in_sizes; (void)n_in; (void)out_size;
  const float* x     = (const float*)d_in[0];
  // d_in[1] = causal mask (int32) -- causality computed analytically, not read
  const float* w_qkv = (const float*)d_in[2];
  const float* b_qkv = (const float*)d_in[3];
  const float* w_out = (const float*)d_in[4];
  const float* b_out = (const float*)d_in[5];
  float* out = (float*)d_out;

  // workspace layout (bf16 = 2B each)
  const size_t szX    = (size_t)kM * kD * 2;        // 16 MB
  const size_t szWqkv = (size_t)3 * kD * kD * 2;    //  6 MB
  const size_t szWout = (size_t)kD * kD * 2;        //  2 MB
  const size_t szQKV  = (size_t)kBH * kT * kHD * 2; // 16 MB each
  const size_t need = szX + szWqkv + szWout + 3 * szQKV + szX;
  if (ws_size < need) return;

  char* ws = (char*)d_ws;
  u16* xbf   = (u16*)ws; ws += szX;
  u16* wqkvT = (u16*)ws; ws += szWqkv;
  u16* woutT = (u16*)ws; ws += szWout;
  u16* qb    = (u16*)ws; ws += szQKV;
  u16* kbuf  = (u16*)ws; ws += szQKV;
  u16* vT    = (u16*)ws; ws += szQKV;
  u16* y     = (u16*)ws; ws += szX;

  pack_bf16_kernel<<<2048, 256, 0, stream>>>(x, xbf, kM * kD);
  transpose_pack_kernel<<<2048, 256, 0, stream>>>(w_qkv, wqkvT, kD, 3 * kD);
  transpose_pack_kernel<<<1024, 256, 0, stream>>>(w_out, woutT, kD, kD);

  dim3 g1(kM / 128, (3 * kD) / 128);   // 64 x 24
  gemm_bf16_kernel<<<g1, 256, 0, stream>>>(xbf, wqkvT, b_qkv, kD, 3 * kD, 0,
                                           qb, kbuf, vT, nullptr);

  dim3 g2(kT / 128, kBH);              // 16 x 64
  flash_attn_kernel<<<g2, 256, 0, stream>>>(qb, kbuf, vT, y);

  dim3 g3(kM / 128, kD / 128);         // 64 x 8
  gemm_bf16_kernel<<<g3, 256, 0, stream>>>(y, woutT, b_out, kD, kD, 1,
                                           nullptr, nullptr, nullptr, out);
}
